// WindowAttention_77257871721042
// MI455X (gfx1250) — matrix operand order repaired and checked by reading, compiled, not run on hardware
//
#include <hip/hip_runtime.h>

typedef __attribute__((ext_vector_type(16))) __bf16        v16bf;
typedef __attribute__((ext_vector_type(8)))  float         v8f;
typedef __attribute__((ext_vector_type(8)))  unsigned int  v8u;
typedef __attribute__((ext_vector_type(4)))  unsigned int  u32x4;
typedef __attribute__((ext_vector_type(8)))  unsigned int  u32x8;

#define DIM     512
#define HEADS   16
#define HD      32
#define NTOK    49
#define BI      2048
#define NWIN    64
#define M_TOTAL (BI * NTOK)     /* 100352 = 784 * 128 */
#define SQRT_HD 5.656854249492381f   /* score = qk / scale, scale = hd^-0.5 */

// ---------- helpers ----------

__device__ __forceinline__ unsigned short f2bf(float f) {
  unsigned int u = __float_as_uint(f);
  u += 0x7FFFu + ((u >> 16) & 1u);           // round-to-nearest-even
  return (unsigned short)(u >> 16);
}

__device__ __forceinline__ v8f wmma_bf16(v8u a, v8u b, v8f c) {
  return __builtin_amdgcn_wmma_f32_16x16x32_bf16(
      false, __builtin_bit_cast(v16bf, a),
      false, __builtin_bit_cast(v16bf, b),
      (short)0, c, false, false);
}

// low 32 bits of a generic shared-memory pointer == LDS byte offset
__device__ __forceinline__ unsigned lds_off(const void* p) {
  return (unsigned)(unsigned long long)p;
}

// Issue one TDM 2D tile load: global (row-major, stride0 elems of 2 bytes)
// -> LDS contiguous tile_d0 x tile_d1. OOB rows/cols are zero-filled.
// Per-wave instruction (EXEC ignored); call from one wave only.
// D# layout per CDNA5 ISA 8.3/8.4; groups 2/3 zeroed (tile_dim2=0 -> 2D).
__device__ __forceinline__ void tdm_load_2d(unsigned lds, const void* gptr,
                                            unsigned tensor_d0, unsigned tensor_d1,
                                            unsigned tile_d0, unsigned tile_d1,
                                            unsigned stride0) {
  const unsigned long long ga = (unsigned long long)gptr;
  u32x4 g0;
  g0[0] = 1u;                                            // count=1, no gather
  g0[1] = lds;                                           // lds_addr (bytes)
  g0[2] = (unsigned)(ga & 0xFFFFFFFFull);                // global_addr[31:0]
  g0[3] = (unsigned)((ga >> 32) & 0x01FFFFFFull) | (2u << 30);  // addr[56:32], type=2
  u32x8 g1;
  g1[0] = 1u << 16;                                      // wg_mask=0, data_size=2B
  g1[1] = (tensor_d0 & 0xFFFFu) << 16;                   // atomic_bar=0 | tdim0.lo
  g1[2] = (tensor_d0 >> 16) | ((tensor_d1 & 0xFFFFu) << 16);
  g1[3] = (tensor_d1 >> 16) | (tile_d0 << 16);
  g1[4] = tile_d1;                                       // tile_dim2 = 0 (2D)
  g1[5] = stride0;                                       // tensor_dim0_stride lo
  g1[6] = 0u;
  g1[7] = 0u;
  u32x4 gz = (u32x4){0u, 0u, 0u, 0u};
  asm volatile("tensor_load_to_lds %0, %1, %2, %3"
               :: "s"(g0), "s"(g1), "s"(gz), "s"(gz)
               : "memory");
}

// A fragment: 16x32 bf16 tile, row-major [m][k] in LDS, ld = row stride (elems).
__device__ __forceinline__ v8u frag_a(const unsigned short* base, int ld) {
  const int lane = threadIdx.x & 31;
  const int m = lane & 15, half = lane >> 4;
  v8u r;
#pragma unroll
  for (int i = 0; i < 8; ++i) {
    const int k = ((i >> 2) << 4) + (half << 3) + ((i & 3) << 1);
    r[i] = *(const unsigned int*)(base + m * ld + k);
  }
  return r;
}

// B fragment: 32x16 bf16 tile stored TRANSPOSED as [n][k], ld = row stride.
__device__ __forceinline__ v8u frag_b(const unsigned short* base, int ld) {
  const int lane = threadIdx.x & 31;
  const int n = lane & 15, kh = lane >> 4;
  v8u r;
#pragma unroll
  for (int v = 0; v < 8; ++v) {
    const int k = ((v >> 2) << 4) + (kh << 3) + ((v & 3) << 1);
    r[v] = *(const unsigned int*)(base + n * ld + k);
  }
  return r;
}

// ---------- kernel 0: x fp32 -> bf16 ----------

__global__ __launch_bounds__(256)
void xconv_kernel(const float* __restrict__ x, unsigned short* __restrict__ xb) {
  const size_t half_total = (size_t)M_TOTAL * DIM / 2;
  const size_t i = (size_t)blockIdx.x * blockDim.x + threadIdx.x;
  if (i < half_total) {
    const float2 t = ((const float2*)x)[i];
    ((unsigned int*)xb)[i] = (unsigned)f2bf(t.x) | ((unsigned)f2bf(t.y) << 16);
  }
}

// ---------- kernel 1: weight transpose/convert + bias densify ----------

__global__ void prep_kernel(const float* __restrict__ qkv_w,
                            const float* __restrict__ proj_w,
                            const float* __restrict__ bias_table,
                            const int*   __restrict__ rel_index,
                            unsigned short* __restrict__ qkv_wT,
                            unsigned short* __restrict__ proj_wT,
                            float* __restrict__ rel_bias) {
  const int T1 = 1536 * 512, T2 = 512 * 512, T3 = HEADS * NTOK * NTOK;
  const int total = T1 + T2 + T3;
  for (int idx = blockIdx.x * blockDim.x + threadIdx.x; idx < total;
       idx += gridDim.x * blockDim.x) {
    if (idx < T1) {
      const int n = idx / 512, k = idx % 512;
      qkv_wT[n * 512 + k] = f2bf(qkv_w[k * 1536 + n]);
    } else if (idx < T1 + T2) {
      const int t = idx - T1;
      const int n = t / 512, k = t % 512;
      proj_wT[n * 512 + k] = f2bf(proj_w[k * 512 + n]);
    } else {
      const int t = idx - T1 - T2;
      const int h = t / (NTOK * NTOK), p = t % (NTOK * NTOK);
      rel_bias[h * NTOK * NTOK + p] = bias_table[rel_index[p] * HEADS + h];
    }
  }
}

// ---------- shared GEMM body: TDM double-buffered bf16 128x128 tiles ----------

__device__ __forceinline__ void gemm_core(const unsigned short* a, // [M][512] bf16
                                          const unsigned short* wT,// [N][512] bf16
                                          int m0, int n0,
                                          unsigned short (*As)[128 * 32],
                                          unsigned short (*Bs)[128 * 32],
                                          v8f acc[4][2]) {
  const int tid = threadIdx.x;
  const int wave = tid >> 5;
  const int wm = (wave & 1) * 64;
  const int wn = (wave >> 1) * 32;
  const bool issuer = (tid < 32);

#pragma unroll
  for (int i = 0; i < 4; ++i)
#pragma unroll
    for (int j = 0; j < 2; ++j) acc[i][j] = (v8f){};

  if (issuer) {
    tdm_load_2d(lds_off(As[0]), a + (size_t)m0 * 512, 512, 128, 32, 128, 512);
    tdm_load_2d(lds_off(Bs[0]), wT + (size_t)n0 * 512, 512, 128, 32, 128, 512);
  }

  for (int ks = 0; ks < 16; ++ks) {
    const int buf = ks & 1;
    if (issuer) __builtin_amdgcn_s_wait_tensorcnt(0);
    __syncthreads();
    if (issuer && (ks + 1 < 16)) {
      const int k1 = (ks + 1) * 32;
      tdm_load_2d(lds_off(As[buf ^ 1]), a + (size_t)m0 * 512 + k1, 512, 128, 32, 128, 512);
      tdm_load_2d(lds_off(Bs[buf ^ 1]), wT + (size_t)n0 * 512 + k1, 512, 128, 32, 128, 512);
    }
    v8u af[4], bfr[2];
#pragma unroll
    for (int mt = 0; mt < 4; ++mt) af[mt] = frag_a(As[buf] + (wm + mt * 16) * 32, 32);
#pragma unroll
    for (int nt = 0; nt < 2; ++nt) bfr[nt] = frag_b(Bs[buf] + (wn + nt * 16) * 32, 32);
#pragma unroll
    for (int mt = 0; mt < 4; ++mt)
#pragma unroll
      for (int nt = 0; nt < 2; ++nt)
        acc[mt][nt] = wmma_bf16(af[mt], bfr[nt], acc[mt][nt]);
  }
}

// ---------- kernel 2: QKV GEMM ----------

__global__ __launch_bounds__(256)
void gemm_qkv_kernel(const unsigned short* __restrict__ xb,
                     const unsigned short* __restrict__ wT,
                     const float* __restrict__ qkv_b,
                     unsigned short* __restrict__ qb,
                     unsigned short* __restrict__ kb,
                     unsigned short* __restrict__ vb) {
  __shared__ unsigned short As[2][128 * 32];
  __shared__ unsigned short Bs[2][128 * 32];
  const int tid = threadIdx.x;
  const int m0 = blockIdx.x * 128;
  const int n0 = blockIdx.y * 128;
  const int wave = tid >> 5;
  const int wm = (wave & 1) * 64;
  const int wn = (wave >> 1) * 32;

  v8f acc[4][2];
  gemm_core(xb, wT, m0, n0, As, Bs, acc);

  // epilogue: +bias, fold 1/scale into q, scatter q/k/v as [b][h][tok][d] bf16
  const int lane = tid & 31;
  const int nn = lane & 15, half = lane >> 4;
#pragma unroll
  for (int mt = 0; mt < 4; ++mt) {
#pragma unroll
    for (int nt = 0; nt < 2; ++nt) {
#pragma unroll
      for (int r = 0; r < 8; ++r) {
        const int m = m0 + wm + mt * 16 + r + half * 8;
        const int n = n0 + wn + nt * 16 + nn;
        float val = acc[mt][nt][r] + qkv_b[n];
        const int which = n >> 9;
        const int rem = n & 511;
        const int h = rem >> 5, d = rem & 31;
        const int b = m / NTOK, tok = m - b * NTOK;
        if (which == 0) val *= SQRT_HD;
        const size_t di = ((size_t)(b * HEADS + h) * NTOK + tok) * HD + d;
        unsigned short* dst = (which == 0) ? qb : (which == 1) ? kb : vb;
        dst[di] = f2bf(val);
      }
    }
  }
}

// ---------- kernel 3: windowed attention, one block per (b, h) ----------

__global__ __launch_bounds__(128)
void attn_kernel(const unsigned short* __restrict__ qb,
                 const unsigned short* __restrict__ kb,
                 const unsigned short* __restrict__ vb,
                 const float* __restrict__ rel_bias,
                 const float* __restrict__ mask,
                 unsigned short* __restrict__ ao) {
  __shared__ unsigned short qs[64 * 32];
  __shared__ unsigned short ks[64 * 32];
  __shared__ unsigned short vts[32 * 64];   // v transposed: [d][key]
  __shared__ float S[64 * 65];
  __shared__ unsigned short P[64 * 64];

  const int tid = threadIdx.x;
  const int bh = blockIdx.x;
  const int b = bh >> 4, h = bh & 15;
  const int w = b & (NWIN - 1);
  const size_t base = (size_t)bh * NTOK * HD;

  // TDM: stage q and k as 64x32 tiles; rows 49..63 zero-filled via OOB
  if (tid < 32) {
    tdm_load_2d(lds_off(qs), qb + base, 32, NTOK, 32, 64, 32);
    tdm_load_2d(lds_off(ks), kb + base, 32, NTOK, 32, 64, 32);
  }
  // stage v transposed manually (zero-padded keys 49..63)
  for (int idx = tid; idx < 32 * 64; idx += 128) {
    const int d = idx >> 6, key = idx & 63;
    vts[idx] = (key < NTOK) ? vb[base + key * HD + d] : (unsigned short)0;
  }
  if (tid < 32) __builtin_amdgcn_s_wait_tensorcnt(0);
  __syncthreads();

  // S = q @ k^T : wave mt owns query tile mt, sweeps 4 key tiles (K=32 = HD)
  {
    const int mt = tid >> 5;
    const int lane = tid & 31;
    const int nn = lane & 15, half = lane >> 4;
    const v8u af = frag_a(qs + mt * 16 * 32, 32);
#pragma unroll
    for (int nt = 0; nt < 4; ++nt) {
      const v8u bfr = frag_b(ks + nt * 16 * 32, 32);
      v8f c = (v8f){};
      c = wmma_bf16(af, bfr, c);
#pragma unroll
      for (int r = 0; r < 8; ++r)
        S[(mt * 16 + r + half * 8) * 65 + nt * 16 + nn] = c[r];
    }
  }
  __syncthreads();

  // softmax over keys with rel-bias + window mask; emit P in bf16 (padded)
  if (tid < NTOK) {
    const int row = tid;
    const float* bias = rel_bias + (h * NTOK + row) * NTOK;
    const float* mrow = mask + ((size_t)w * NTOK + row) * NTOK;
    float* srow = S + row * 65;
    float mx = -3.4e38f;
    for (int m = 0; m < NTOK; ++m) {
      const float v = srow[m] * SQRT_HD + bias[m] + mrow[m];
      srow[m] = v;
      mx = fmaxf(mx, v);
    }
    float sum = 0.f;
    for (int m = 0; m < NTOK; ++m) {
      const float e = __expf(srow[m] - mx);
      srow[m] = e;
      sum += e;
    }
    const float inv = 1.f / sum;
    unsigned short* prow = P + row * 64;
    for (int m = 0; m < NTOK; ++m) prow[m] = f2bf(srow[m] * inv);
    for (int m = NTOK; m < 64; ++m) prow[m] = 0;
  } else if (tid < 64) {
    unsigned short* prow = P + tid * 64;
    for (int m = 0; m < 64; ++m) prow[m] = 0;
  }
  __syncthreads();

  // out = P @ V : wave mt owns query tile mt; nt sweeps d tiles; K=64 in 2 steps
  {
    const int mt = tid >> 5;
    const int lane = tid & 31;
    const int nn = lane & 15, half = lane >> 4;
#pragma unroll
    for (int nt = 0; nt < 2; ++nt) {
      v8f c = (v8f){};
#pragma unroll
      for (int kk = 0; kk < 2; ++kk) {
        const v8u af = frag_a(P + mt * 16 * 64 + kk * 32, 64);
        const v8u bfr = frag_b(vts + nt * 16 * 64 + kk * 32, 64);
        c = wmma_bf16(af, bfr, c);
      }
#pragma unroll
      for (int r = 0; r < 8; ++r) {
        const int mq = mt * 16 + r + half * 8;
        if (mq < NTOK) {
          const int d = nt * 16 + nn;
          ao[((size_t)b * NTOK + mq) * DIM + h * HD + d] = f2bf(c[r]);
        }
      }
    }
  }
}

// ---------- kernel 4: output projection GEMM (fp32 out) ----------

__global__ __launch_bounds__(256)
void gemm_proj_kernel(const unsigned short* __restrict__ a,
                      const unsigned short* __restrict__ wT,
                      const float* __restrict__ proj_b,
                      float* __restrict__ out) {
  __shared__ unsigned short As[2][128 * 32];
  __shared__ unsigned short Bs[2][128 * 32];
  const int tid = threadIdx.x;
  const int m0 = blockIdx.x * 128;
  const int n0 = blockIdx.y * 128;
  const int wave = tid >> 5;
  const int wm = (wave & 1) * 64;
  const int wn = (wave >> 1) * 32;

  v8f acc[4][2];
  gemm_core(a, wT, m0, n0, As, Bs, acc);

  const int lane = tid & 31;
  const int nn = lane & 15, half = lane >> 4;
#pragma unroll
  for (int mt = 0; mt < 4; ++mt) {
#pragma unroll
    for (int nt = 0; nt < 2; ++nt) {
#pragma unroll
      for (int r = 0; r < 8; ++r) {
        const int m = m0 + wm + mt * 16 + r + half * 8;
        const int n = n0 + wn + nt * 16 + nn;
        out[(size_t)m * 512 + n] = acc[mt][nt][r] + proj_b[n];
      }
    }
  }
}

// ---------- launch ----------

extern "C" void kernel_launch(void* const* d_in, const int* in_sizes, int n_in,
                              void* d_out, int out_size, void* d_ws, size_t ws_size,
                              hipStream_t stream) {
  const float* x          = (const float*)d_in[0];
  const float* mask       = (const float*)d_in[1];
  const float* qkv_w      = (const float*)d_in[2];
  const float* qkv_b      = (const float*)d_in[3];
  const float* proj_w     = (const float*)d_in[4];
  const float* proj_b     = (const float*)d_in[5];
  const float* bias_table = (const float*)d_in[6];
  const int*   rel_index  = (const int*)d_in[7];

  char* ws = (char*)d_ws;
  size_t off = 0;
  auto alloc = [&](size_t bytes) -> void* {
    void* p = ws + off;
    off = (off + bytes + 255) & ~(size_t)255;
    return p;
  };
  unsigned short* qkv_wT  = (unsigned short*)alloc((size_t)1536 * 512 * 2);
  unsigned short* proj_wT = (unsigned short*)alloc((size_t)512 * 512 * 2);
  float*          relb    = (float*)alloc((size_t)HEADS * NTOK * NTOK * 4);
  unsigned short* xb      = (unsigned short*)alloc((size_t)M_TOTAL * DIM * 2);
  const size_t qkv_elems  = (size_t)BI * HEADS * NTOK * HD;
  unsigned short* qb = (unsigned short*)alloc(qkv_elems * 2);
  unsigned short* kb = (unsigned short*)alloc(qkv_elems * 2);
  unsigned short* vb = (unsigned short*)alloc(qkv_elems * 2);
  unsigned short* ao = (unsigned short*)alloc((size_t)M_TOTAL * DIM * 2);

  xconv_kernel<<<(M_TOTAL * (DIM / 2) + 255) / 256, 256, 0, stream>>>(x, xb);

  prep_kernel<<<512, 256, 0, stream>>>(qkv_w, proj_w, bias_table, rel_index,
                                       qkv_wT, proj_wT, relb);

  dim3 g1(M_TOTAL / 128, 1536 / 128);
  gemm_qkv_kernel<<<g1, 256, 0, stream>>>(xb, qkv_wT, qkv_b, qb, kb, vb);

  attn_kernel<<<BI * HEADS, 128, 0, stream>>>(qb, kb, vb, relb, mask, ao);

  dim3 g2(M_TOTAL / 128, 512 / 128);
  gemm_proj_kernel<<<g2, 256, 0, stream>>>(ao, proj_wT, proj_b, (float*)d_out);
}
